// HyenaMixer1D_52544629899325
// MI455X (gfx1250) — compile-verified, matrix operand order checked by reading, not run on hardware
//
#include <hip/hip_runtime.h>
#include <hip/hip_bf16.h>

typedef __attribute__((ext_vector_type(16))) __bf16 v16bf;
typedef __attribute__((ext_vector_type(8)))  float  v8f;
typedef __attribute__((ext_vector_type(8)))  __bf16 bf16x8;
typedef __attribute__((ext_vector_type(4)))  __bf16 bf16x4;

#define BATCH 4
#define TLEN  4096
#define HDIM  1024
#define KTAPS 256

// ---------------------------------------------------------------------------
// fp32 -> bf16 conversion (vectorized, grid-stride)
// ---------------------------------------------------------------------------
__global__ __launch_bounds__(256) void f32_to_bf16_kernel(const float* __restrict__ src,
                                                          __bf16* __restrict__ dst,
                                                          long n4) {
  long i = (long)blockIdx.x * blockDim.x + threadIdx.x;
  long stride = (long)gridDim.x * blockDim.x;
  for (; i < n4; i += stride) {
    float4 v = ((const float4*)src)[i];
    bf16x4 o;
    o[0] = (__bf16)v.x; o[1] = (__bf16)v.y; o[2] = (__bf16)v.z; o[3] = (__bf16)v.w;
    ((bf16x4*)dst)[i] = o;
  }
}

// ---------------------------------------------------------------------------
// bf16 WMMA GEMM, fully compile-time configured:
//   C[m,n] = sum_k A(m,k)*B(k,n) + bias
//   A element (m,k) at A[m*SAM + k*SAK]   (A_K_CONTIG => SAK==1, else SAM==1)
//   B element (k,n) at B[n*SBN + k]       (k contiguous)
//   C element (m,n) at C[m*SCM + n]
// Block tile 128(M) x 128(N), K stepped by 32, double-buffered LDS with
// register prefetch of the next K-tile: one barrier per K-step.
// 256 threads = 8 wave32 in 4x2 grid; each wave owns 32x64 = 2x4 WMMA tiles.
// ---------------------------------------------------------------------------
template <bool OUT_BF16, bool BIAS_PER_M, bool A_K_CONTIG, int KD,
          int SAM, int SAK, int SBN, int SCM, long BSA, long BSB, long BSC>
__global__ __launch_bounds__(256) void wmma_gemm_bf16(const __bf16* __restrict__ A,
                                                      const __bf16* __restrict__ Bm,
                                                      void* __restrict__ Cm,
                                                      const float* __restrict__ bias) {
  __shared__ __bf16 sA[2][128][40];   // [buf][m][k], padded rows (80B, 16B aligned)
  __shared__ __bf16 sB[2][128][40];   // [buf][n][k]

  const int tid = threadIdx.x;
  const int m0  = blockIdx.y * 128;
  const int n0  = blockIdx.x * 128;
  const int bz  = blockIdx.z;

  const __bf16* Ab = A  + (long)bz * BSA;
  const __bf16* Bb = Bm + (long)bz * BSB;

  const int lane = tid & 31;
  const int wid  = tid >> 5;
  const int wrow = wid >> 1;           // 0..3 -> M offset wrow*32
  const int wcol = wid & 1;            // 0..1 -> N offset wcol*64
  const int l16  = lane & 15;
  const int khi  = (lane >> 4) << 3;   // 0 (lanes 0-15) or 8 (lanes 16-31)

  // per-thread staging coordinates (two 8-element chunks each for A and B)
  int a_m[2], a_kc[2], a_k[2], a_mb[2], b_n[2], b_kc[2];
#pragma unroll
  for (int c0 = 0; c0 < 2; ++c0) {
    int c = tid + c0 * 256;
    a_m[c0]  = c >> 2;        a_kc[c0] = (c & 3) << 3;   // A k-contig path
    a_k[c0]  = c & 31;        a_mb[c0] = (c >> 5) << 3;  // A m-contig path
    b_n[c0]  = c >> 2;        b_kc[c0] = (c & 3) << 3;
  }

  v8f acc[2][4] = {};

  constexpr int NT = KD / 32;          // number of K tiles

  // ---- prologue: stage K-tile 0 into buffer 0 ----
#pragma unroll
  for (int c0 = 0; c0 < 2; ++c0) {
    if (A_K_CONTIG) {
      *(bf16x8*)&sA[0][a_m[c0]][a_kc[c0]] =
          *(const bf16x8*)(Ab + (long)(m0 + a_m[c0]) * SAM + a_kc[c0]);
    } else {
      bf16x8 v = *(const bf16x8*)(Ab + (long)a_k[c0] * SAK + (m0 + a_mb[c0]));
#pragma unroll
      for (int j = 0; j < 8; ++j) sA[0][a_mb[c0] + j][a_k[c0]] = v[j];
    }
    *(bf16x8*)&sB[0][b_n[c0]][b_kc[c0]] =
        *(const bf16x8*)(Bb + (long)(n0 + b_n[c0]) * SBN + b_kc[c0]);
  }
  __syncthreads();

#pragma unroll 1
  for (int it = 0; it < NT - 1; ++it) {
    const int cur = it & 1;
    const int nxt = cur ^ 1;
    const int kk2 = (it + 1) * 32;

    // ---- build fragments from current buffer (2x ds_load_b128 each) ----
    v16bf af[2], bfr[4];
#pragma unroll
    for (int i = 0; i < 2; ++i) {
      int mrow = wrow * 32 + i * 16 + l16;
      bf16x8 lo = *(const bf16x8*)&sA[cur][mrow][khi];
      bf16x8 hi = *(const bf16x8*)&sA[cur][mrow][16 + khi];
      af[i] = __builtin_shufflevector(lo, hi, 0,1,2,3,4,5,6,7,8,9,10,11,12,13,14,15);
    }
#pragma unroll
    for (int j = 0; j < 4; ++j) {
      int ncol = wcol * 64 + j * 16 + l16;
      bf16x8 lo = *(const bf16x8*)&sB[cur][ncol][khi];
      bf16x8 hi = *(const bf16x8*)&sB[cur][ncol][16 + khi];
      bfr[j] = __builtin_shufflevector(lo, hi, 0,1,2,3,4,5,6,7,8,9,10,11,12,13,14,15);
    }

    // ---- prefetch next K-tile into registers (overlaps with WMMA) ----
    bf16x8 pra[2], prb[2];
#pragma unroll
    for (int c0 = 0; c0 < 2; ++c0) {
      if (A_K_CONTIG)
        pra[c0] = *(const bf16x8*)(Ab + (long)(m0 + a_m[c0]) * SAM + (kk2 + a_kc[c0]));
      else
        pra[c0] = *(const bf16x8*)(Ab + (long)(kk2 + a_k[c0]) * SAK + (m0 + a_mb[c0]));
      prb[c0] = *(const bf16x8*)(Bb + (long)(n0 + b_n[c0]) * SBN + (kk2 + b_kc[c0]));
    }

    // ---- 8 WMMA ops ----
#pragma unroll
    for (int i = 0; i < 2; ++i)
#pragma unroll
      for (int j = 0; j < 4; ++j)
        acc[i][j] = __builtin_amdgcn_wmma_f32_16x16x32_bf16(
            false, af[i], false, bfr[j], (short)0, acc[i][j], false, false);

    // ---- commit prefetched tile into the other buffer (no fence needed:
    //      everyone finished reading it at the previous barrier) ----
#pragma unroll
    for (int c0 = 0; c0 < 2; ++c0) {
      if (A_K_CONTIG) {
        *(bf16x8*)&sA[nxt][a_m[c0]][a_kc[c0]] = pra[c0];
      } else {
#pragma unroll
        for (int j = 0; j < 8; ++j) sA[nxt][a_mb[c0] + j][a_k[c0]] = pra[c0][j];
      }
      *(bf16x8*)&sB[nxt][b_n[c0]][b_kc[c0]] = prb[c0];
    }
    __syncthreads();
  }

  // ---- final K-tile: fragments + WMMA only ----
  {
    const int cur = (NT - 1) & 1;
    v16bf af[2], bfr[4];
#pragma unroll
    for (int i = 0; i < 2; ++i) {
      int mrow = wrow * 32 + i * 16 + l16;
      bf16x8 lo = *(const bf16x8*)&sA[cur][mrow][khi];
      bf16x8 hi = *(const bf16x8*)&sA[cur][mrow][16 + khi];
      af[i] = __builtin_shufflevector(lo, hi, 0,1,2,3,4,5,6,7,8,9,10,11,12,13,14,15);
    }
#pragma unroll
    for (int j = 0; j < 4; ++j) {
      int ncol = wcol * 64 + j * 16 + l16;
      bf16x8 lo = *(const bf16x8*)&sB[cur][ncol][khi];
      bf16x8 hi = *(const bf16x8*)&sB[cur][ncol][16 + khi];
      bfr[j] = __builtin_shufflevector(lo, hi, 0,1,2,3,4,5,6,7,8,9,10,11,12,13,14,15);
    }
#pragma unroll
    for (int i = 0; i < 2; ++i)
#pragma unroll
      for (int j = 0; j < 4; ++j)
        acc[i][j] = __builtin_amdgcn_wmma_f32_16x16x32_bf16(
            false, af[i], false, bfr[j], (short)0, acc[i][j], false, false);
  }

  // ---- epilogue: C/D layout, VGPR r -> M = r (+8 for hi lanes), N = lane%16 ----
  const int mhalf = (lane >> 4) << 3;
#pragma unroll
  for (int i = 0; i < 2; ++i) {
#pragma unroll
    for (int j = 0; j < 4; ++j) {
      int nn = n0 + wcol * 64 + j * 16 + l16;
#pragma unroll
      for (int r = 0; r < 8; ++r) {
        int mm = m0 + wrow * 32 + i * 16 + r + mhalf;
        float v = acc[i][j][r] + (BIAS_PER_M ? bias[mm] : bias[nn]);
        long off = (long)bz * BSC + (long)mm * SCM + nn;
        if (OUT_BF16) ((__bf16*)Cm)[off] = (__bf16)v;
        else          ((float*)Cm)[off]  = v;
      }
    }
  }
}

// ---------------------------------------------------------------------------
// Fused causal depthwise conv (256 taps) + sigmoid cross-gating.
// One block per (batch, channel). Entire za/zb rows + halo staged in LDS,
// then 'cat' written IN PLACE over z (safe: barrier after full row load).
// ---------------------------------------------------------------------------
__global__ __launch_bounds__(256) void dwconv_gate_kernel(const __bf16* __restrict__ z,
                                                          __bf16* __restrict__ cat,
                                                          const float* __restrict__ wA,
                                                          const float* __restrict__ bA,
                                                          const float* __restrict__ wB,
                                                          const float* __restrict__ bB) {
  __shared__ float swa[KTAPS];
  __shared__ float swb[KTAPS];
  __shared__ __bf16 sza[KTAPS + TLEN];   // data at offset 256; [1..255] = zero halo
  __shared__ __bf16 szb[KTAPS + TLEN];

  const int tid = threadIdx.x;
  const int b = blockIdx.x / HDIM;
  const int h = blockIdx.x % HDIM;

  const long rowA = (long)b * 2 * HDIM + h;
  const long rowB = rowA + HDIM;
  const __bf16* za = z + rowA * TLEN;
  const __bf16* zb = z + rowB * TLEN;

  swa[tid] = wA[h * KTAPS + tid];
  swb[tid] = wB[h * KTAPS + tid];
  sza[tid] = (__bf16)0.0f;             // zero halo (indices 0..255)
  szb[tid] = (__bf16)0.0f;
#pragma unroll
  for (int c0 = 0; c0 < 2; ++c0) {
    int off = (tid + c0 * 256) * 8;
    *(bf16x8*)&sza[KTAPS + off] = *(const bf16x8*)(za + off);
    *(bf16x8*)&szb[KTAPS + off] = *(const bf16x8*)(zb + off);
  }
  __syncthreads();

  const float bav = bA[h];
  const float bbv = bB[h];
  __bf16* ca_out = cat + rowA * TLEN;
  __bf16* cb_out = cat + rowB * TLEN;

#pragma unroll 1
  for (int s = 0; s < TLEN / 256; ++s) {
    int t = s * 256 + tid;
    const __bf16* pa = &sza[t + 1];    // covers z[t-255 .. t]
    const __bf16* pb = &szb[t + 1];
    float aa = 0.f, ab = 0.f;
#pragma unroll 8
    for (int j = 0; j < KTAPS; ++j) {
      aa += (float)pa[j] * swa[j];
      ab += (float)pb[j] * swb[j];
    }
    float ca = aa + bav;
    float cb = ab + bbv;
    float ga = ca * (1.0f / (1.0f + __expf(-cb)));
    float gb = cb * (1.0f / (1.0f + __expf(-ca)));
    ca_out[t] = (__bf16)ga;
    cb_out[t] = (__bf16)gb;
  }
}

// ---------------------------------------------------------------------------
extern "C" void kernel_launch(void* const* d_in, const int* in_sizes, int n_in,
                              void* d_out, int out_size, void* d_ws, size_t ws_size,
                              hipStream_t stream) {
  const float* x     = (const float*)d_in[0];
  const float* w_in  = (const float*)d_in[1];
  const float* b_in  = (const float*)d_in[2];
  const float* w_a   = (const float*)d_in[3];
  const float* b_a   = (const float*)d_in[4];
  const float* w_b   = (const float*)d_in[5];
  const float* b_b   = (const float*)d_in[6];
  const float* w_out = (const float*)d_in[7];
  const float* b_out = (const float*)d_in[8];
  float* out = (float*)d_out;

  // workspace layout (bf16)
  __bf16* x_bf     = (__bf16*)d_ws;                         // B*T*H
  __bf16* w_in_bf  = x_bf + (long)BATCH * TLEN * HDIM;      // 2H*H
  __bf16* w_out_bf = w_in_bf + 2L * HDIM * HDIM;            // H*2H
  __bf16* z_bf     = w_out_bf + 2L * HDIM * HDIM;           // B*2H*T (cat in-place)

  // 1) demote GEMM operands to bf16
  f32_to_bf16_kernel<<<2048, 256, 0, stream>>>(x,     x_bf,     (long)BATCH * TLEN * HDIM / 4);
  f32_to_bf16_kernel<<<512,  256, 0, stream>>>(w_in,  w_in_bf,  2L * HDIM * HDIM / 4);
  f32_to_bf16_kernel<<<512,  256, 0, stream>>>(w_out, w_out_bf, 2L * HDIM * HDIM / 4);

  // 2) GEMM1: z[b,e,t] = sum_h w_in[e,h] * x[b,t,h] + b_in[e]
  //    M=2048(e), N=4096(t), K=1024(h); bias per-M; bf16 output
  {
    dim3 grid(TLEN / 128, 2 * HDIM / 128, BATCH);
    wmma_gemm_bf16<true, true, /*A_K_CONTIG*/true, /*KD*/HDIM,
                   /*SAM*/HDIM, /*SAK*/1, /*SBN*/HDIM, /*SCM*/TLEN,
                   /*BSA*/0L, /*BSB*/(long)TLEN * HDIM, /*BSC*/2L * HDIM * TLEN>
        <<<grid, 256, 0, stream>>>(w_in_bf, x_bf, z_bf, b_in);
  }

  // 3) depthwise causal conv + gating, cat written in place over z
  dwconv_gate_kernel<<<BATCH * HDIM, 256, 0, stream>>>(z_bf, z_bf, w_a, b_a, w_b, b_b);

  // 4) GEMM2: out[b,t,h] = sum_e cat[b,e,t] * w_out[h,e] + b_out[h]
  //    M=4096(t), N=1024(h), K=2048(e); bias per-N; fp32 output
  {
    dim3 grid(HDIM / 128, TLEN / 128, BATCH);
    wmma_gemm_bf16<false, false, /*A_K_CONTIG*/false, /*KD*/2 * HDIM,
                   /*SAM*/1, /*SAK*/TLEN, /*SBN*/2 * HDIM, /*SCM*/HDIM,
                   /*BSA*/2L * HDIM * TLEN, /*BSB*/0L, /*BSC*/(long)TLEN * HDIM>
        <<<grid, 256, 0, stream>>>(z_bf, w_out_bf, out, b_out);
  }
}
